// CopyCatDecoder_54056458387569
// MI455X (gfx1250) — matrix operand order repaired and checked by reading, not compiled, not run on hardware
//
#include <hip/hip_runtime.h>
#include <stdint.h>

// ---------------------------------------------------------------------------
// CopyCatDecoder on MI455X (gfx1250): bf16 WMMA GEMMs + fused softmax/scatter.
// Shapes: T=128, B=8, S=256, D=512, V=50257, L=2.
// Dominant work: 1024x50257x512 GEMM -> v_wmma_f32_16x16x32_bf16 with
// on-the-fly f32->bf16 weight packing via v_perm_b32.
// ---------------------------------------------------------------------------

typedef __attribute__((ext_vector_type(16))) __bf16   v16bf;
typedef __attribute__((ext_vector_type(8)))  __bf16   v8bf;
typedef __attribute__((ext_vector_type(8)))  float    v8f;
typedef __attribute__((ext_vector_type(8)))  unsigned v8u;

static constexpr int   T_ = 128;
static constexpr int   B_ = 8;
static constexpr int   S_ = 256;
static constexpr int   D_ = 512;
static constexpr int   L_ = 2;
static constexpr long  V_ = 50257;

__device__ __forceinline__ __bf16 f2bf(float f) {
  unsigned u = __builtin_bit_cast(unsigned, f);
  unsigned r = u + 0x7fffu + ((u >> 16) & 1u);   // round-to-nearest-even
  unsigned short h = (unsigned short)(r >> 16);
  return __builtin_bit_cast(__bf16, h);
}

// Pack two f32 into one dword of two bf16 (round-half-up) with v_perm_b32.
__device__ __forceinline__ unsigned pk_bf(float a, float b) {
  unsigned ua = __builtin_bit_cast(unsigned, a) + 0x8000u;
  unsigned ub = __builtin_bit_cast(unsigned, b) + 0x8000u;
  // combined = {src0=ub (bytes 4..7), src1=ua (bytes 0..3)}
  // result = {ub[31:16], ua[31:16]}  -> (lo=a, hi=b)
  return __builtin_amdgcn_perm(ub, ua, 0x07060302u);
}

__device__ __forceinline__ v8bf ld8bf(const __bf16* p) {
  return __builtin_bit_cast(v8bf, *reinterpret_cast<const uint4*>(p));
}

// ---------------------------------------------------------------------------
// Wave-level GEMM: C[M,N] = alpha * A[M,K] * B^T  (B given as N rows of K)
// 1 wave (32 lanes) per block computes a 16 x (NT*16) tile (NT accumulators).
// A: bf16, row stride A_rs (elements), batch stride A_batch (z = blockIdx.z).
// B: bf16 or f32 (BF32), row n holds K contiguous elements, stride B_rs.
// C index: z*C_batch + (m/divB)*C_sT + (m%divB)*C_sB + n  (supports the
// (t,b)->(b,t,V) remap of the vocab GEMM; divB=1 for plain layouts).
// ---------------------------------------------------------------------------
template <bool BF32, int NT>
__global__ __launch_bounds__(32) void k_gemm(
    const __bf16* __restrict__ A, long A_batch, long A_rs,
    const void*   __restrict__ Bv, long B_batch, long B_rs,
    float* __restrict__ Cf, __bf16* __restrict__ C16,
    const float* __restrict__ bias,
    long C_batch, int divB, long C_sT, long C_sB,
    int N, int K, float alpha, int addC) {
  const int ln = threadIdx.x;
  const int lm = ln & 15;      // column-in-tile / row-in-tile (mod 16)
  const int h  = ln >> 4;      // lane group
  const int m0 = blockIdx.y << 4;
  const int n0 = blockIdx.x * (NT * 16);
  const long zb = blockIdx.z;

  A += zb * A_batch;
  const __bf16* Bb = nullptr;
  const float*  Bf = nullptr;
  if constexpr (BF32) Bf = (const float*)Bv + zb * B_batch;
  else                Bb = (const __bf16*)Bv + zb * B_batch;

  long brow[NT];
#pragma unroll
  for (int j = 0; j < NT; ++j) {
    int n = n0 + j * 16 + lm;
    brow[j] = (long)((n < N) ? n : (N - 1)) * B_rs;  // clamp edge (loads only)
  }
  const __bf16* arow = A + (long)(m0 + lm) * A_rs;

  v8f acc[NT] = {};
  for (int k0 = 0; k0 < K; k0 += 32) {
    // A fragment: lane (m=lm, group h) holds K = h*16 .. h*16+15
    v8bf alo = ld8bf(arow + k0 + h * 16);
    v8bf ahi = ld8bf(arow + k0 + h * 16 + 8);
    v16bf af = __builtin_shufflevector(alo, ahi, 0, 1, 2, 3, 4, 5, 6, 7, 8, 9,
                                       10, 11, 12, 13, 14, 15);
#pragma unroll
    for (int j = 0; j < NT; ++j) {
      // B fragment: lane column n=lm, lane group h holds K = h*16 .. h*16+15
      v16bf bfr;
      if constexpr (BF32) {
        const float* bp = Bf + brow[j] + k0 + h * 16;
        const float4* q4 = reinterpret_cast<const float4*>(bp);
        float4 x0 = q4[0], x1 = q4[1], x2 = q4[2], x3 = q4[3];
        v8u pk;
        pk[0] = pk_bf(x0.x, x0.y);
        pk[1] = pk_bf(x0.z, x0.w);
        pk[2] = pk_bf(x1.x, x1.y);
        pk[3] = pk_bf(x1.z, x1.w);
        pk[4] = pk_bf(x2.x, x2.y);
        pk[5] = pk_bf(x2.z, x2.w);
        pk[6] = pk_bf(x3.x, x3.y);
        pk[7] = pk_bf(x3.z, x3.w);
        bfr = __builtin_bit_cast(v16bf, pk);
      } else {
        const __bf16* bp = Bb + brow[j] + k0 + h * 16;
        bfr = __builtin_shufflevector(ld8bf(bp), ld8bf(bp + 8), 0, 1, 2, 3, 4,
                                      5, 6, 7, 8, 9, 10, 11, 12, 13, 14, 15);
      }
      acc[j] = __builtin_amdgcn_wmma_f32_16x16x32_bf16(
          false, af, false, bfr, (short)0, acc[j], false, false);
    }
  }

  // Epilogue: C layout -> VGPR r holds row m = m0 + h*8 + r, col n0+j*16+lm.
#pragma unroll
  for (int j = 0; j < NT; ++j) {
    int n = n0 + j * 16 + lm;
    if (n >= N) continue;
    float bv = bias ? bias[n] : 0.0f;
#pragma unroll
    for (int r = 0; r < 8; ++r) {
      int m = m0 + h * 8 + r;
      float v = acc[j][r] * alpha + bv;
      long ci = zb * C_batch + (long)(m / divB) * C_sT +
                (long)(m % divB) * C_sB + n;
      if (addC) v += Cf[ci];
      if (Cf)  Cf[ci]  = v;
      if (C16) C16[ci] = f2bf(v);
    }
  }
}

// ---------------------------------------------------------------------------
// Row softmax over S=256 (blockDim==S, one row per block, row = b*T + t).
// Optional: accumulate into attn_acc, emit bf16 probs, scale by (1-pr_gen).
// ---------------------------------------------------------------------------
__global__ __launch_bounds__(256) void k_softmax(
    const float* __restrict__ Sin, float* __restrict__ Pf,
    __bf16* __restrict__ P16, float* __restrict__ Acc,
    const float* __restrict__ scaleRow, const unsigned char* __restrict__ mask,
    int T, int Slen) {
  const int row = blockIdx.x;
  const int b = row / T;
  const int s = threadIdx.x;
  float x = Sin[(long)row * Slen + s];
  if (mask[b * Slen + s]) x = -1.0e9f;

  __shared__ float red[8];
  float m = x;
#pragma unroll
  for (int o = 16; o > 0; o >>= 1) m = fmaxf(m, __shfl_xor(m, o, 32));
  const int w = threadIdx.x >> 5;
  if ((threadIdx.x & 31) == 0) red[w] = m;
  __syncthreads();
  float mx = red[0];
#pragma unroll
  for (int i = 1; i < 8; ++i) mx = fmaxf(mx, red[i]);

  float e = __expf(x - mx);
  float sum = e;
#pragma unroll
  for (int o = 16; o > 0; o >>= 1) sum += __shfl_xor(sum, o, 32);
  __syncthreads();
  if ((threadIdx.x & 31) == 0) red[w] = sum;
  __syncthreads();
  float tot = 0.0f;
#pragma unroll
  for (int i = 0; i < 8; ++i) tot += red[i];

  float p = e / tot;
  if (scaleRow) p *= (1.0f - scaleRow[row]);
  const long idx = (long)row * Slen + s;
  if (Pf)  Pf[idx] = p;
  if (P16) P16[idx] = f2bf(p);
  if (Acc) Acc[idx] += p;
}

// pr_gen[b*T+t] = sigmoid(dot(out[t,b,:], w) + b0); one wave per row.
__global__ __launch_bounds__(256) void k_prgen(
    const float* __restrict__ outf, const float* __restrict__ w,
    const float* __restrict__ b0, float* __restrict__ pr, int T, int B,
    int D) {
  const int wv = threadIdx.x >> 5, ln = threadIdx.x & 31;
  const int row = blockIdx.x * 8 + wv;   // b*T + t
  const int b = row / T, t = row % T;
  const float* x = outf + ((long)t * B + b) * D;
  float s = 0.0f;
  for (int d = ln; d < D; d += 32) s += x[d] * w[d];
#pragma unroll
  for (int o = 16; o > 0; o >>= 1) s += __shfl_xor(s, o, 32);
  if (ln == 0) pr[row] = 1.0f / (1.0f + __expf(-(s + b0[0])));
}

// Online (max, sumexp) per logits row of length V.
__global__ __launch_bounds__(256) void k_rowstats(
    const float* __restrict__ logits, float* __restrict__ st, long V) {
  const long row = blockIdx.x;
  const float* x = logits + row * V;
  float m = -3.4e38f, s = 0.0f;
  for (long v = threadIdx.x; v < V; v += 256) {
    float xv = x[v];
    float nm = fmaxf(m, xv);
    s = s * __expf(m - nm) + __expf(xv - nm);
    m = nm;
  }
  __shared__ float sm[256], ss[256];
  sm[threadIdx.x] = m;
  ss[threadIdx.x] = s;
  __syncthreads();
  for (int o = 128; o > 0; o >>= 1) {
    if (threadIdx.x < (unsigned)o) {
      float m2 = sm[threadIdx.x + o], s2 = ss[threadIdx.x + o];
      float nm = fmaxf(sm[threadIdx.x], m2);
      ss[threadIdx.x] =
          ss[threadIdx.x] * __expf(sm[threadIdx.x] - nm) + s2 * __expf(m2 - nm);
      sm[threadIdx.x] = nm;
    }
    __syncthreads();
  }
  if (threadIdx.x == 0) { st[row * 2] = sm[0]; st[row * 2 + 1] = ss[0]; }
}

// gen_probs = softmax(logits)*pr_gen, then scatter-add copy_mass at ids.
__global__ __launch_bounds__(256) void k_finalize(
    float* __restrict__ out, const float* __restrict__ st,
    const float* __restrict__ pr, const float* __restrict__ cmass,
    const int* __restrict__ ids, int T, int S, long V) {
  const long row = blockIdx.x;   // b*T + t  (matches (b,t,V) row order)
  const float m = st[row * 2];
  const float scale = pr[row] / st[row * 2 + 1];
  float* x = out + row * V;
  for (long v = threadIdx.x; v < V; v += 256) x[v] = __expf(x[v] - m) * scale;
  __threadfence();
  __syncthreads();
  const int b = (int)(row / T);
  const int s = threadIdx.x;     // blockDim == S == 256
  const int id = ids[b * S + s];
  atomicAdd(&x[id], cmass[row * S + s]);
}

// coverage_reg = mean over (b,s) of -log(clip(sum_t attn_acc / L, eps, 1)).
__global__ void k_cover(const float* __restrict__ acc,
                        float* __restrict__ outScalar, int B, int T, int S,
                        float invL, float invBS) {
  const int i = blockIdx.x * blockDim.x + threadIdx.x;
  if (i >= B * S) return;
  const int b = i / S, s = i % S;
  float a = 0.0f;
  for (int t = 0; t < T; ++t) a += acc[((long)b * T + t) * S + s];
  a *= invL;
  a = fminf(fmaxf(a, 1e-9f), 1.0f);
  atomicAdd(outScalar, -__logf(a) * invBS);
}

// ------------------------- prep / utility kernels --------------------------
__global__ void k_cvt_copy(const float* __restrict__ src,
                           float* __restrict__ df, __bf16* __restrict__ db,
                           long n) {
  long i = (long)blockIdx.x * blockDim.x + threadIdx.x;
  if (i < n) {
    float v = src[i];
    if (df) df[i] = v;
    db[i] = f2bf(v);
  }
}

__global__ void k_transpose_gm(const float* __restrict__ gm,
                               __bf16* __restrict__ gmT, int S, int B, int D) {
  long i = (long)blockIdx.x * blockDim.x + threadIdx.x;
  long n = (long)S * B * D;
  if (i >= n) return;
  int s = (int)(i / (B * D));
  int r = (int)(i % (B * D));
  int b = r / D, d = r % D;
  gmT[((long)b * D + d) * S + s] = f2bf(gm[i]);
}

__global__ void k_cvt_wT(const float* __restrict__ W, __bf16* __restrict__ WT,
                         int L, int D) {
  long i = (long)blockIdx.x * blockDim.x + threadIdx.x;
  long n = (long)L * D * D;
  if (i >= n) return;
  int l = (int)(i / ((long)D * D));
  int r = (int)(i % ((long)D * D));
  int d = r / D, e = r % D;
  WT[(long)l * D * D + (long)e * D + d] = f2bf(W[i]);
}

__global__ void k_zero(float* __restrict__ p, long n) {
  long i = (long)blockIdx.x * blockDim.x + threadIdx.x;
  if (i < n) p[i] = 0.0f;
}

__global__ void k_init_scalars(float* __restrict__ p) {
  if (threadIdx.x < 2) p[threadIdx.x] = 0.0f;
}

// ---------------------------------------------------------------------------
extern "C" void kernel_launch(void* const* d_in, const int* in_sizes, int n_in,
                              void* d_out, int out_size, void* d_ws,
                              size_t ws_size, hipStream_t stream) {
  (void)in_sizes; (void)n_in; (void)out_size; (void)ws_size;
  const float* target = (const float*)d_in[0];            // (T,B,D)
  const float* gm     = (const float*)d_in[1];            // (S,B,D)
  const int*   ids    = (const int*)d_in[2];              // (B,S)
  const unsigned char* mask = (const unsigned char*)d_in[3];  // (B,S) bool
  const float* Wq = (const float*)d_in[4];                // (L,D,D)
  const float* Wo = (const float*)d_in[5];                // (L,D,D)
  const float* genW = (const float*)d_in[6];              // (V,D)
  const float* genB = (const float*)d_in[7];              // (V,)
  const float* wgw  = (const float*)d_in[8];              // (1,D)
  const float* wgb  = (const float*)d_in[9];              // (1,)
  float* out = (float*)d_out;

  // --- scratch layout (needs ~15.2 MB) ---
  char* ws = (char*)d_ws;
  size_t o = 0;
  auto alloc = [&](size_t bytes) {
    char* p = ws + o;
    o = (o + bytes + 255) & ~(size_t)255;
    return p;
  };
  const long TBD = (long)T_ * B_ * D_;      // 524288
  const long SBD = (long)S_ * B_ * D_;      // 1048576
  const long BTS = (long)B_ * T_ * S_;      // 262144
  const long LDD = (long)L_ * D_ * D_;      // 524288
  float*  OUT_F32  = (float*)alloc(TBD * 4);
  __bf16* OUT_B16  = (__bf16*)alloc(TBD * 2);
  __bf16* Q_B16    = (__bf16*)alloc(TBD * 2);
  __bf16* GM_B16   = (__bf16*)alloc(SBD * 2);
  __bf16* GMT_B16  = (__bf16*)alloc(SBD * 2);
  __bf16* WQT_B16  = (__bf16*)alloc(LDD * 2);
  __bf16* WOT_B16  = (__bf16*)alloc(LDD * 2);
  float*  SCORES   = (float*)alloc(BTS * 4);
  __bf16* A_B16    = (__bf16*)alloc(BTS * 2);
  float*  ATTN     = (float*)alloc(BTS * 4);
  __bf16* CTX_B16  = (__bf16*)alloc(TBD * 2);
  float*  COPYM    = (float*)alloc(BTS * 4);
  float*  PRGEN    = (float*)alloc((long)B_ * T_ * 4);
  float*  ROWSTATS = (float*)alloc((long)B_ * T_ * 2 * 4);

  const long NBTV = (long)B_ * T_ * V_;
  const float rsqd = 0.044194173824159216f;   // 1/sqrt(512)

  // --- prep ---
  k_cvt_copy<<<(unsigned)((TBD + 255) / 256), 256, 0, stream>>>(
      target, OUT_F32, OUT_B16, TBD);
  k_cvt_copy<<<(unsigned)((SBD + 255) / 256), 256, 0, stream>>>(
      gm, nullptr, GM_B16, SBD);
  k_transpose_gm<<<(unsigned)((SBD + 255) / 256), 256, 0, stream>>>(
      gm, GMT_B16, S_, B_, D_);
  k_cvt_wT<<<(unsigned)((LDD + 255) / 256), 256, 0, stream>>>(Wq, WQT_B16, L_, D_);
  k_cvt_wT<<<(unsigned)((LDD + 255) / 256), 256, 0, stream>>>(Wo, WOT_B16, L_, D_);
  k_zero<<<(unsigned)((BTS + 255) / 256), 256, 0, stream>>>(ATTN, BTS);
  k_init_scalars<<<1, 32, 0, stream>>>(out + NBTV);

  // --- decoder layers ---
  for (int l = 0; l < L_; ++l) {
    // q = (out @ Wq[l]) * 1/sqrt(d)   [M=1024, N=512, K=512]
    k_gemm<false, 4><<<dim3(D_ / 64, (T_ * B_) / 16, 1), 32, 0, stream>>>(
        OUT_B16, 0, D_, (const void*)(WQT_B16 + (long)l * D_ * D_), 0, D_,
        nullptr, Q_B16, nullptr, 0, 1, D_, 0, D_, D_, rsqd, 0);
    // scores[b,t,s] = q_b @ gm_b^T     [per batch: M=128, N=256, K=512]
    k_gemm<false, 4><<<dim3(S_ / 64, T_ / 16, B_), 32, 0, stream>>>(
        Q_B16, D_, (long)B_ * D_, (const void*)GM_B16, D_, (long)B_ * D_,
        SCORES, nullptr, nullptr, (long)T_ * S_, 1, S_, 0, S_, D_, 1.0f, 0);
    // softmax over s; accumulate attn_acc; emit bf16 probs
    k_softmax<<<B_ * T_, S_, 0, stream>>>(SCORES, nullptr, A_B16, ATTN,
                                          nullptr, mask, T_, S_);
    // ctx[t,b,d] = a_b @ gm_b          [per batch: M=128, N=512, K=256]
    k_gemm<false, 4><<<dim3(D_ / 64, T_ / 16, B_), 32, 0, stream>>>(
        A_B16, (long)T_ * S_, S_, (const void*)GMT_B16, (long)D_ * S_, S_,
        nullptr, CTX_B16, nullptr, D_, 1, (long)B_ * D_, 0, D_, S_, 1.0f, 0);
    // out += ctx @ Wo[l]               [M=1024, N=512, K=512], residual add
    k_gemm<false, 4><<<dim3(D_ / 64, (T_ * B_) / 16, 1), 32, 0, stream>>>(
        CTX_B16, 0, D_, (const void*)(WOT_B16 + (long)l * D_ * D_), 0, D_,
        OUT_F32, OUT_B16, nullptr, 0, 1, D_, 0, D_, D_, 1.0f, 1);
  }

  // --- pointer-generator head ---
  k_prgen<<<(B_ * T_) / 8, 256, 0, stream>>>(OUT_F32, wgw, wgb, PRGEN, T_, B_, D_);
  // a_g scores = intermediate @ gm^T, softmax, fuse (1-pr_gen) -> copy_mass
  k_gemm<false, 4><<<dim3(S_ / 64, T_ / 16, B_), 32, 0, stream>>>(
      OUT_B16, D_, (long)B_ * D_, (const void*)GM_B16, D_, (long)B_ * D_,
      SCORES, nullptr, nullptr, (long)T_ * S_, 1, S_, 0, S_, D_, 1.0f, 0);
  k_softmax<<<B_ * T_, S_, 0, stream>>>(SCORES, COPYM, nullptr, nullptr, PRGEN,
                                        mask, T_, S_);

  // vocab logits: out_bf16 @ genW^T + bias -> d_out as (b,t,V)
  // 16x128 tiles (NT=8); m = t*B + b -> index = b*T*V + t*V + v
  k_gemm<true, 8><<<dim3((unsigned)((V_ + 127) / 128), (T_ * B_) / 16, 1), 32,
                    0, stream>>>(
      OUT_B16, 0, D_, (const void*)genW, 0, D_, out, nullptr, genB, 0, B_, V_,
      (long)T_ * V_, (int)V_, D_, 1.0f, 0);

  // softmax over V (online stats), scale by pr_gen, scatter copy mass
  k_rowstats<<<B_ * T_, 256, 0, stream>>>(out, ROWSTATS, V_);
  k_finalize<<<B_ * T_, 256, 0, stream>>>(out, ROWSTATS, PRGEN, COPYM, ids, T_,
                                          S_, V_);

  // coverage scalar (total_cov_pen already zeroed)
  k_cover<<<(B_ * S_ + 255) / 256, 256, 0, stream>>>(
      ATTN, out + NBTV, B_, T_, S_, 1.0f / L_, 1.0f / (B_ * S_));
}
